// GaussianKernel2_84524956385427
// MI455X (gfx1250) — compile-verified, matrix operand order checked by reading, not run on hardware
//
#include <hip/hip_runtime.h>

// ---------------------------------------------------------------------------
// RBF kernel: out = exp(-gamma * (||x||^2 - 2 x@L^T + ||l||^2))
// B=16384, L=2048, F=1024 (fp32).  Compute-bound (AI ~343 FLOP/B).
// GEMM on WMMA bf16 pipes with hi/lo split precision (3 bf16 WMMAs ~ fp32).
// Round 3: 64x64 wave tiles (4x4 WMMA accs, 48 WMMA / 32 ds_load per step)
// so LDS read bandwidth no longer outpaces the matrix pipes.
// ---------------------------------------------------------------------------

typedef __attribute__((ext_vector_type(16))) __bf16          v16bf;
typedef __attribute__((ext_vector_type(8)))  float           v8f;
typedef __attribute__((ext_vector_type(8)))  unsigned short  u16x8;
typedef __attribute__((ext_vector_type(4)))  unsigned short  u16x4;
typedef __attribute__((ext_vector_type(4)))  float           f32x4;

#define KDIM   1024     // feature dim F (fixed by reference)
#define TILE_M 256
#define TILE_N 128
#define BK     32
#define LSTR   40       // padded LDS row stride (bf16 elems): rows 80B apart ->
                        // 16B-aligned b128 frag loads, 20-bank stride = conflict-free

static __device__ __forceinline__ unsigned short f32_to_bf16_rne(float f) {
  unsigned u = __builtin_bit_cast(unsigned, f);
  u += 0x7FFFu + ((u >> 16) & 1u);          // round-to-nearest-even
  return (unsigned short)(u >> 16);
}
static __device__ __forceinline__ float bf16_bits_to_f32(unsigned short h) {
  return __builtin_bit_cast(float, (unsigned)h << 16);
}

union Frag {            // 16 bf16 per lane = 8 VGPRs (ISA 16-bit A/B layout)
  v16bf bf;
  u16x8 h[2];
};

// --------------------------- row-norm pre-pass -----------------------------
__global__ __launch_bounds__(256)
void rownorm_kernel(const float* __restrict__ src, float* __restrict__ dst) {
  const int row = blockIdx.x;
  const int t   = threadIdx.x;
  float s = 0.f;
  #pragma unroll
  for (int i = 0; i < KDIM / 256; ++i) {
    float v = src[(size_t)row * KDIM + t + i * 256];
    s += v * v;
  }
  #pragma unroll
  for (int off = 16; off > 0; off >>= 1) s += __shfl_xor(s, off, 32);
  __shared__ float red[8];
  if ((t & 31) == 0) red[t >> 5] = s;
  __syncthreads();
  if (t == 0) {
    float tot = 0.f;
    #pragma unroll
    for (int w = 0; w < 8; ++w) tot += red[w];
    dst[row] = tot;
  }
}

// ----------------------- fp32 -> bf16 hi/lo pre-pass -----------------------
__global__ __launch_bounds__(256)
void convert_bf16_kernel(const float* __restrict__ src,
                         unsigned short* __restrict__ hi,
                         unsigned short* __restrict__ lo, long n4) {
  long i = (long)blockIdx.x * 256 + threadIdx.x;   // in float4 units
  if (i >= n4) return;
  f32x4 v = ((const f32x4*)src)[i];
  u16x4 h, l;
  #pragma unroll
  for (int e = 0; e < 4; ++e) {
    unsigned short hb = f32_to_bf16_rne(v[e]);
    h[e] = hb;
    l[e] = f32_to_bf16_rne(v[e] - bf16_bits_to_f32(hb));
  }
  ((u16x4*)hi)[i] = h;
  ((u16x4*)lo)[i] = l;
}

// ------------------------------ fused GEMM ---------------------------------
// PRE=true : P0..P3 = xhi, xlo, lhi, llo (bf16 bits, pre-converted in ws)
// PRE=false: P0 = X (fp32), P1 = Lm (fp32); convert in-kernel (fallback)
template <bool PRE>
__global__ __launch_bounds__(256)
void rbf_wmma_kernel(const void* __restrict__ P0, const void* __restrict__ P1,
                     const void* __restrict__ P2, const void* __restrict__ P3,
                     const float* __restrict__ gammaRaw,
                     const float* __restrict__ xsq, const float* __restrict__ lsq,
                     float* __restrict__ out, int N) {
  __shared__ __align__(16) unsigned short sXhi[TILE_M * LSTR];
  __shared__ __align__(16) unsigned short sXlo[TILE_M * LSTR];
  __shared__ __align__(16) unsigned short sLhi[TILE_N * LSTR];
  __shared__ __align__(16) unsigned short sLlo[TILE_N * LSTR];
  // total: (256+256+128+128)*40*2B = 61,440 B (< 64 KB static cap)

  const int tid   = threadIdx.x;
  const int lane  = tid & 31;
  const int wave  = tid >> 5;
  const int waveM = wave >> 1;            // 0..3 : 64-row patch of 256
  const int waveN = wave & 1;             // 0..1 : 64-col patch of 128
  const int m0    = blockIdx.y * TILE_M;
  const int n0    = blockIdx.x * TILE_N;

  // ---------------- global->reg staging, reg->LDS store --------------------
  // PRE path: pure bf16 b128 copies (no VALU conversion in the hot loop).
  u16x8 sxh[4], sxl[4], slh[2], sll[2];   // PRE staging
  f32x4 xr[8], lr[4];                     // !PRE staging

  auto gload = [&](int k0) {
    if constexpr (PRE) {
      const unsigned short* xhi = (const unsigned short*)P0;
      const unsigned short* xlo = (const unsigned short*)P1;
      const unsigned short* lhi = (const unsigned short*)P2;
      const unsigned short* llo = (const unsigned short*)P3;
      #pragma unroll
      for (int it = 0; it < 4; ++it) {    // X panel: 256x32 = 1024 u16x8 chunks
        int flat = it * 256 + tid;
        int r = flat >> 2, c = (flat & 3) * 8;
        size_t off = (size_t)(m0 + r) * KDIM + k0 + c;
        sxh[it] = *(const u16x8*)(xhi + off);
        sxl[it] = *(const u16x8*)(xlo + off);
      }
      #pragma unroll
      for (int it = 0; it < 2; ++it) {    // L panel: 128x32 = 512 u16x8 chunks
        int flat = it * 256 + tid;
        int r = flat >> 2, c = (flat & 3) * 8;
        size_t off = (size_t)(n0 + r) * KDIM + k0 + c;
        slh[it] = *(const u16x8*)(lhi + off);
        sll[it] = *(const u16x8*)(llo + off);
      }
    } else {
      const float* X  = (const float*)P0;
      const float* Lm = (const float*)P1;
      #pragma unroll
      for (int it = 0; it < 8; ++it) {    // X panel: 256x32 = 2048 float4
        int flat = it * 256 + tid;
        int r = flat >> 3, c = (flat & 7) * 4;
        xr[it] = *(const f32x4*)(X + (size_t)(m0 + r) * KDIM + k0 + c);
      }
      #pragma unroll
      for (int it = 0; it < 4; ++it) {    // L panel: 128x32 = 1024 float4
        int flat = it * 256 + tid;
        int r = flat >> 3, c = (flat & 7) * 4;
        lr[it] = *(const f32x4*)(Lm + (size_t)(n0 + r) * KDIM + k0 + c);
      }
    }
  };

  auto lstore = [&]() {
    if constexpr (PRE) {
      #pragma unroll
      for (int it = 0; it < 4; ++it) {
        int flat = it * 256 + tid;
        int r = flat >> 2, c = (flat & 3) * 8;
        *(u16x8*)(sXhi + r * LSTR + c) = sxh[it];
        *(u16x8*)(sXlo + r * LSTR + c) = sxl[it];
      }
      #pragma unroll
      for (int it = 0; it < 2; ++it) {
        int flat = it * 256 + tid;
        int r = flat >> 2, c = (flat & 3) * 8;
        *(u16x8*)(sLhi + r * LSTR + c) = slh[it];
        *(u16x8*)(sLlo + r * LSTR + c) = sll[it];
      }
    } else {
      #pragma unroll
      for (int it = 0; it < 8; ++it) {
        int flat = it * 256 + tid;
        int r = flat >> 3, c = (flat & 7) * 4;
        u16x4 h, l;
        #pragma unroll
        for (int e = 0; e < 4; ++e) {
          float v = xr[it][e];
          unsigned short hb = f32_to_bf16_rne(v);
          h[e] = hb;
          l[e] = f32_to_bf16_rne(v - bf16_bits_to_f32(hb));
        }
        *(u16x4*)(sXhi + r * LSTR + c) = h;
        *(u16x4*)(sXlo + r * LSTR + c) = l;
      }
      #pragma unroll
      for (int it = 0; it < 4; ++it) {
        int flat = it * 256 + tid;
        int r = flat >> 3, c = (flat & 7) * 4;
        u16x4 h, l;
        #pragma unroll
        for (int e = 0; e < 4; ++e) {
          float v = lr[it][e];
          unsigned short hb = f32_to_bf16_rne(v);
          h[e] = hb;
          l[e] = f32_to_bf16_rne(v - bf16_bits_to_f32(hb));
        }
        *(u16x4*)(sLhi + r * LSTR + c) = h;
        *(u16x4*)(sLlo + r * LSTR + c) = l;
      }
    }
  };

  // ISA 16-bit A/B fragment layout: lanes 0-15 -> K{0..7,16..23} of row M=lane,
  // lanes 16-31 -> K{8..15,24..31} of row M=lane-16.  Two b128 LDS loads/frag.
  const int fragRow = lane & 15;
  const int kc0     = (lane >> 4) * 8;
  auto ldFrag = [&](const unsigned short* base, int row) {
    Frag f;
    f.h[0] = *(const u16x8*)(base + row * LSTR + kc0);
    f.h[1] = *(const u16x8*)(base + row * LSTR + kc0 + 16);
    return f;
  };

  v8f acc[4][4];
  #pragma unroll
  for (int ti = 0; ti < 4; ++ti)
    #pragma unroll
    for (int tj = 0; tj < 4; ++tj)
      acc[ti][tj] = (v8f){0.f, 0.f, 0.f, 0.f, 0.f, 0.f, 0.f, 0.f};

  gload(0);
  lstore();
  __syncthreads();

  const int KSTEPS = KDIM / BK;           // 32
  for (int ks = 0; ks < KSTEPS; ++ks) {
    if (ks + 1 < KSTEPS) gload((ks + 1) * BK);   // hide VMEM under WMMA

    Frag bHi[4], bLo[4];
    #pragma unroll
    for (int tj = 0; tj < 4; ++tj) {
      int rr = waveN * 64 + tj * 16 + fragRow;
      bHi[tj] = ldFrag(sLhi, rr);
      bLo[tj] = ldFrag(sLlo, rr);
    }
    #pragma unroll
    for (int ti = 0; ti < 4; ++ti) {
      int rr = waveM * 64 + ti * 16 + fragRow;
      Frag aHi = ldFrag(sXhi, rr);
      Frag aLo = ldFrag(sXlo, rr);
      #pragma unroll
      for (int tj = 0; tj < 4; ++tj) {
        // split-precision: dot ~= hi*hi + hi*lo + lo*hi  (drop lo*lo)
        acc[ti][tj] = __builtin_amdgcn_wmma_f32_16x16x32_bf16(
            false, aHi.bf, false, bHi[tj].bf, (short)0, acc[ti][tj], false, false);
        acc[ti][tj] = __builtin_amdgcn_wmma_f32_16x16x32_bf16(
            false, aHi.bf, false, bLo[tj].bf, (short)0, acc[ti][tj], false, false);
        acc[ti][tj] = __builtin_amdgcn_wmma_f32_16x16x32_bf16(
            false, aLo.bf, false, bHi[tj].bf, (short)0, acc[ti][tj], false, false);
      }
    }
    __syncthreads();                      // all waves done reading LDS
    if (ks + 1 < KSTEPS) lstore();
    __syncthreads();                      // new panel visible
  }

  // ---- epilogue: d2 = xsq + lsq - 2*dot ; out = exp(-gamma*d2) ------------
  // C/D layout: VGPR j of lane l -> M = j + 8*(l>=16), N = l&15
  const float negGamma = -__expf(gammaRaw[0]);
  const int mBase = m0 + waveM * 64 + ((lane >> 4) << 3);
  const int nBase = n0 + waveN * 64 + (lane & 15);
  #pragma unroll
  for (int ti = 0; ti < 4; ++ti) {
    #pragma unroll
    for (int tj = 0; tj < 4; ++tj) {
      int n = nBase + tj * 16;
      float lq = lsq[n];
      #pragma unroll
      for (int j = 0; j < 8; ++j) {
        int m = mBase + ti * 16 + j;
        float d2 = xsq[m] + lq - 2.0f * acc[ti][tj][j];
        // write-once 128MB stream: keep it out of L2 (X/L panels live there)
        __builtin_nontemporal_store(__expf(negGamma * d2), out + (size_t)m * N + n);
      }
    }
  }
}

// ------------------------------- launcher ----------------------------------
extern "C" void kernel_launch(void* const* d_in, const int* in_sizes, int n_in,
                              void* d_out, int out_size, void* d_ws, size_t ws_size,
                              hipStream_t stream) {
  const float* X  = (const float*)d_in[0];   // [B, F] fp32
  const float* Lm = (const float*)d_in[1];   // [L, F] fp32
  const float* g  = (const float*)d_in[2];   // [1]    fp32 (raw gamma param)
  float* out = (float*)d_out;                // [B, L] fp32

  const int B = in_sizes[0] / KDIM;          // 16384
  const int L = in_sizes[1] / KDIM;          // 2048
  const size_t nx = (size_t)B * KDIM;
  const size_t nl = (size_t)L * KDIM;

  float* xsq = (float*)d_ws;                 // [B]
  float* lsq = xsq + B;                      // [L]
  const size_t normBytes = (size_t)(B + L) * sizeof(float);   // 72 KB
  const size_t needPre   = normBytes + (nx + nl) * 2 * sizeof(unsigned short);

  rownorm_kernel<<<B, 256, 0, stream>>>(X, xsq);
  rownorm_kernel<<<L, 256, 0, stream>>>(Lm, lsq);

  dim3 grid(L / TILE_N, B / TILE_M);         // 16 x 64 tiles

  if (ws_size >= needPre) {
    // pre-convert once: GEMM hot loop becomes pure b128 copy + WMMA
    unsigned short* xhi = (unsigned short*)(lsq + L);
    unsigned short* xlo = xhi + nx;
    unsigned short* lhi = xlo + nx;
    unsigned short* llo = lhi + nl;
    convert_bf16_kernel<<<(int)(nx / 4 / 256), 256, 0, stream>>>(X,  xhi, xlo, (long)(nx / 4));
    convert_bf16_kernel<<<(int)(nl / 4 / 256), 256, 0, stream>>>(Lm, lhi, llo, (long)(nl / 4));
    rbf_wmma_kernel<true><<<grid, 256, 0, stream>>>(xhi, xlo, lhi, llo,
                                                    g, xsq, lsq, out, L);
  } else {
    // fallback: convert inside the GEMM kernel
    rbf_wmma_kernel<false><<<grid, 256, 0, stream>>>(X, Lm, nullptr, nullptr,
                                                     g, xsq, lsq, out, L);
  }
}
